// Conv2DSpatial_59940563583417
// MI455X (gfx1250) — matrix operand order repaired and checked by reading, not compiled, not run on hardware
//
#include <hip/hip_runtime.h>

// CDNA5 / gfx1250 spatial-RNN, transposed GEMM orientation:
//   t'(:,pix) = relu( W^T * t'(:,pix_shifted) ),  acc = x + sum t_k
// A-matrix = W^T (register bf16 fragments, loaded from a transposed bf16 LDS
//            copy built by one rolled cooperative loop),
// B-matrix = activation tile from LDS (lane = pixel, K = channel, contiguous),
// D-matrix = lane holds 8 consecutive out-channels of one pixel.
// relu = single v_med3_num_f32; bf16 pair-pack = single v_perm_b32.

typedef __attribute__((ext_vector_type(16))) __bf16          v16bf;
typedef __attribute__((ext_vector_type(8)))  float           v8f;
typedef __attribute__((ext_vector_type(16))) unsigned short  v16u;
typedef __attribute__((ext_vector_type(8)))  unsigned short  v8u;
typedef __attribute__((ext_vector_type(4)))  unsigned int    v4ui;
typedef __attribute__((ext_vector_type(4)))  float           v4f;

#define BATCH    8
#define HW       192
#define CCH      64
#define RSTEPS   8
#define LPITCH   (HW + 2)      // 1 zero guard pixel each end
#define NTHREADS 128
#define PT_PER_WAVE 3          // 4 waves * 3 pixel-tiles * 16 = 192 px
#define FMAXF    3.402823466e+38f

// one-time (W) conversion: round-to-nearest-even
__device__ __forceinline__ unsigned short f32_bf16_rne(float f) {
  unsigned u = __float_as_uint(f);
  u += 0x7FFFu + ((u >> 16) & 1u);
  return (unsigned short)(u >> 16);
}

// relu as one v_med3_num_f32
__device__ __forceinline__ float relu1(float v) {
  return __builtin_amdgcn_fmed3f(v, 0.0f, FMAXF);
}

// truncate-pack two f32 -> one dword of bf16 pair, single v_perm_b32
__device__ __forceinline__ unsigned pack_bf16(float lo, float hi) {
  return __builtin_amdgcn_perm(__float_as_uint(hi), __float_as_uint(lo), 0x07060302u);
}

// 16 consecutive bf16 from LDS -> v16bf (two aligned ds_load_b128)
__device__ __forceinline__ v16bf load16bf(const unsigned short* p) {
  const v8u lo = *(const v8u*)p;
  const v8u hi = *(const v8u*)(p + 8);
  v16u t;
#pragma unroll
  for (int j = 0; j < 8; ++j) { t[j] = lo[j]; t[j + 8] = hi[j]; }
  return __builtin_bit_cast(v16bf, t);
}

__global__ __launch_bounds__(NTHREADS, 1)
void spatial_rnn_kernel(const float* __restrict__ x,
                        const float* __restrict__ W0, const float* __restrict__ W1,
                        const float* __restrict__ W2, const float* __restrict__ W3,
                        float* __restrict__ out) {
  __shared__ __align__(16) unsigned short sbuf[2][LPITCH * CCH];

  const int tid  = threadIdx.x;
  const int wave = tid >> 5;
  const int lane = tid & 31;
  const int lhal = lane & 15;
  const int hsel = (lane >> 4) & 1;

  const int line = blockIdx.x;   // row (horiz) or column (vert)
  const int b    = blockIdx.y;
  const int dir  = blockIdx.z;   // 0:left 1:right 2:up 3:down

  const bool horiz = (dir < 2);
  const int  s     = ((dir == 0) || (dir == 2)) ? -1 : +1;   // shift tap
  const float* Wm  = (dir == 0) ? W0 : (dir == 1) ? W1 : (dir == 2) ? W2 : W3;

  long xbase, xpstride;
  if (horiz) { xbase = (long)(b * HW + line) * HW * CCH;            xpstride = CCH; }
  else       { xbase = (long)b * HW * HW * CCH + (long)line * CCH;  xpstride = (long)HW * CCH; }

  long obase, opstride;
  if (horiz) { obase = (long)(b * HW + line) * HW * (4 * CCH) + dir * CCH; opstride = 4 * CCH; }
  else       { obase = (long)b * HW * HW * (4 * CCH) + (long)line * (4 * CCH) + dir * CCH;
               opstride = (long)HW * 4 * CCH; }

  // ---- cooperative build of W^T as bf16 in LDS: wt[m*64 + k] = bf16(W[k][m]).
  //      Rolled loop: coalesced global reads, tiny static footprint. ----
  unsigned short* wt = &sbuf[1][0];            // 64*64*2 = 8 KB (aliased, see below)
  for (int i = tid; i < CCH * CCH; i += NTHREADS) {
    const int k = i >> 6, m = i & 63;
    wt[m * CCH + k] = f32_bf16_rne(Wm[i]);
  }
  __syncthreads();

  // ---- A fragments = W^T (16x32 bf16): lane l: M = l%16;
  //      K offsets {0-15} (lanes<16) / {16-31}; 2 b128 loads each ----
  v16bf Af[2][4];
#pragma unroll
  for (int ks = 0; ks < 2; ++ks) {
#pragma unroll
    for (int ot = 0; ot < 4; ++ot) {
      const unsigned short* wr = wt + (ot * 16 + lhal) * CCH + ks * 32 + hsel * 16;
      const v8u lo = *(const v8u*)wr;          // K off 0-7  (or 16-23)
      const v8u hi = *(const v8u*)(wr + 8);    // K off 8-15 (or 24-31)
      v16u t;
#pragma unroll
      for (int j = 0; j < 8; ++j) { t[j] = lo[j]; t[j + 8] = hi[j]; }
      Af[ks][ot] = __builtin_bit_cast(v16bf, t);
    }
  }
  __syncthreads();   // wt region is overwritten below as ping-pong buffer 1

  // ---- acc = x (f32); seed t0 = x (bf16) into sbuf[0] ----
  float acc[PT_PER_WAVE][4][8];
  const int p0w = wave * (PT_PER_WAVE * 16);
#pragma unroll
  for (int pt = 0; pt < PT_PER_WAVE; ++pt) {
    const int pix = p0w + pt * 16 + lhal;
#pragma unroll
    for (int ot = 0; ot < 4; ++ot) {
      const int ch = ot * 16 + hsel * 8;             // 8 consecutive channels
      const float* xp = x + xbase + (long)pix * xpstride + ch;
      const v4f x0 = *(const v4f*)xp;
      const v4f x1 = *(const v4f*)(xp + 4);
#pragma unroll
      for (int r = 0; r < 4; ++r) { acc[pt][ot][r] = x0[r]; acc[pt][ot][r + 4] = x1[r]; }
      v4ui pk;
      pk[0] = pack_bf16(x0[0], x0[1]); pk[1] = pack_bf16(x0[2], x0[3]);
      pk[2] = pack_bf16(x1[0], x1[1]); pk[3] = pack_bf16(x1[2], x1[3]);
      *(v4ui*)&sbuf[0][(pix + 1) * CCH + ch] = pk;
    }
  }
  // zero guard pixels in both ping-pong buffers
  for (int i = tid; i < 2 * 2 * CCH; i += NTHREADS) {
    const int bsel = i >> 7;
    const int gp   = ((i >> 6) & 1) ? (HW + 1) : 0;
    sbuf[bsel][gp * CCH + (i & 63)] = 0;
  }
  __syncthreads();

  // ---- recurrence: t_k = relu(W^T @ shift(t_{k-1})); acc += t_k ----
  int rb = 0;
  for (int k = 0; k < RSTEPS; ++k) {
    const int wb = rb ^ 1;
    const unsigned short* rbuf = &sbuf[rb][0];
    unsigned short*       wbuf = &sbuf[wb][0];
#pragma unroll
    for (int pt = 0; pt < PT_PER_WAVE; ++pt) {
      const int pb = p0w + pt * 16 + lhal + 1 + s;   // shifted buffer index
      const int cb = hsel * 16;
      const v16bf B0 = load16bf(rbuf + pb * CCH + cb);
      const v16bf B1 = load16bf(rbuf + pb * CCH + 32 + cb);
      const int wb_addr = (p0w + pt * 16 + lhal + 1) * CCH;
#pragma unroll
      for (int ot = 0; ot < 4; ++ot) {
        v8f c = {};
        c = __builtin_amdgcn_wmma_f32_16x16x32_bf16(false, Af[0][ot], false, B0,
                                                    (short)0, c, false, false);
        c = __builtin_amdgcn_wmma_f32_16x16x32_bf16(false, Af[1][ot], false, B1,
                                                    (short)0, c, false, false);
        float v[8];
#pragma unroll
        for (int r = 0; r < 8; ++r) {
          v[r] = relu1(c[r]);                  // one v_med3 per element
          acc[pt][ot][r] += v[r];              // pairs fuse to v_pk_add_f32
        }
        v4ui pk;
        pk[0] = pack_bf16(v[0], v[1]); pk[1] = pack_bf16(v[2], v[3]);
        pk[2] = pack_bf16(v[4], v[5]); pk[3] = pack_bf16(v[6], v[7]);
        *(v4ui*)&wbuf[wb_addr + ot * 16 + hsel * 8] = pk;   // one ds_store_b128
      }
    }
    __syncthreads();   // step k reads precede step k+1 writes
    rb = wb;
  }

  // ---- store acc (f32) to out channel block dir*64; vector b128 stores ----
#pragma unroll
  for (int pt = 0; pt < PT_PER_WAVE; ++pt) {
    const int pix = p0w + pt * 16 + lhal;
#pragma unroll
    for (int ot = 0; ot < 4; ++ot) {
      float* op = out + obase + (long)pix * opstride + ot * 16 + hsel * 8;
      v4f o0, o1;
#pragma unroll
      for (int r = 0; r < 4; ++r) { o0[r] = acc[pt][ot][r]; o1[r] = acc[pt][ot][r + 4]; }
      *(v4f*)op       = o0;
      *(v4f*)(op + 4) = o1;
    }
  }
}

extern "C" void kernel_launch(void* const* d_in, const int* in_sizes, int n_in,
                              void* d_out, int out_size, void* d_ws, size_t ws_size,
                              hipStream_t stream) {
  const float* x  = (const float*)d_in[0];
  const float* Wl = (const float*)d_in[1];
  const float* Wr = (const float*)d_in[2];
  const float* Wu = (const float*)d_in[3];
  const float* Wd = (const float*)d_in[4];
  float* out = (float*)d_out;

  dim3 grid(HW, BATCH, 4);   // line x batch x direction
  dim3 block(NTHREADS);
  hipLaunchKernelGGL(spatial_rnn_kernel, grid, block, 0, stream,
                     x, Wl, Wr, Wu, Wd, out);
}
